// RefinementDecoder_6803228197397
// MI455X (gfx1250) — compile-verified, matrix operand order checked by reading, not run on hardware
//
#include <hip/hip_runtime.h>
#include <hip/hip_bf16.h>
#include <cstdint>

// ---------------- model constants ----------------
#define LSEQ     1024
#define DMODEL   512
#define DINNER   1024
#define DSTATE   64
#define HEADDIM  64
#define NHEADS   16
#define CONVDIM  1152      // DINNER + 2*DSTATE
#define DINPROJ  2192      // 2*DINNER + 2*DSTATE + NHEADS
#define DCONV    4
#define DEPTH    4
#define VOCAB    32000
#define EPSV     1e-5f
#define L2       (2 * LSEQ)   // both directions batched

typedef __bf16 bf16_t;
typedef __attribute__((ext_vector_type(16))) __bf16 v16bf;
typedef __attribute__((ext_vector_type(8)))  __bf16 v8bf;
typedef __attribute__((ext_vector_type(4)))  __bf16 v4bf;
typedef __attribute__((ext_vector_type(8)))  float  v8f;

__device__ __forceinline__ float silu_f(float x) {
    return x / (1.0f + __expf(-x));
}

// -------------------------------------------------------------------------
// f32 -> bf16 conversion, 4-wide (weights converted per call; graph-safe)
// -------------------------------------------------------------------------
__global__ void k_f2bf(const float* __restrict__ s, bf16_t* __restrict__ d, int n) {
    int i = (blockIdx.x * 256 + threadIdx.x) * 4;
    if (i + 3 < n) {
        float4 v = *(const float4*)(s + i);
        v4bf o;
        o[0] = (bf16_t)v.x; o[1] = (bf16_t)v.y; o[2] = (bf16_t)v.z; o[3] = (bf16_t)v.w;
        *(v4bf*)(d + i) = o;
    } else {
        for (; i < n; ++i) d[i] = (bf16_t)s[i];
    }
}

// -------------------------------------------------------------------------
// token embed gather * sqrt(DMODEL); rows [0,L) = fwd, rows [L,2L) = bwd
// (bwd stored time-reversed so both streams run the same causal pipeline)
// -------------------------------------------------------------------------
__global__ void k_embed(const int* __restrict__ tok, const float* __restrict__ EW,
                        float* __restrict__ xall) {
    int t = blockIdx.x;
    int d = threadIdx.x;                 // blockDim.x == DMODEL
    float v = EW[(size_t)tok[t] * DMODEL + d] * 22.62741699796952f;
    xall[(size_t)t * DMODEL + d] = v;
    xall[(size_t)(LSEQ + (LSEQ - 1 - t)) * DMODEL + d] = v;
}

// -------------------------------------------------------------------------
// row layernorm (D cols) -> bf16; per-direction params selected by row block
// -------------------------------------------------------------------------
__global__ void k_layernorm_bf16(const float* __restrict__ x,
                                 const float* __restrict__ w0, const float* __restrict__ b0,
                                 const float* __restrict__ w1, const float* __restrict__ b1,
                                 bf16_t* __restrict__ o, int D) {
    int t = blockIdx.x;
    const float* w = (t >= LSEQ) ? w1 : w0;
    const float* b = (t >= LSEQ) ? b1 : b0;
    const float* xr = x + (size_t)t * D;
    __shared__ float sred[18];
    float s = 0.f, s2 = 0.f;
    for (int i = threadIdx.x; i < D; i += 256) {
        float v = xr[i]; s += v; s2 += v * v;
    }
    for (int off = 16; off; off >>= 1) {
        s  += __shfl_down(s,  off, 32);
        s2 += __shfl_down(s2, off, 32);
    }
    int wid = threadIdx.x >> 5;
    if ((threadIdx.x & 31) == 0) { sred[wid] = s; sred[8 + wid] = s2; }
    __syncthreads();
    if (threadIdx.x == 0) {
        float a = 0.f, c = 0.f;
        for (int i = 0; i < 8; ++i) { a += sred[i]; c += sred[8 + i]; }
        float m = a / (float)D;
        float var = c / (float)D - m * m;
        sred[16] = m;
        sred[17] = rsqrtf(var + EPSV);
    }
    __syncthreads();
    float m = sred[16], inv = sred[17];
    for (int i = threadIdx.x; i < D; i += 256) {
        float v = (xr[i] - m) * inv * w[i] + b[i];
        o[(size_t)t * D + i] = (bf16_t)v;
    }
}

// -------------------------------------------------------------------------
// WMMA bf16 GEMM:  C[M,N](f32) = A[M,K](bf16,row-major) x W[N,K](bf16)^T
//  - block = 256 thr = 8 waves arranged 4(M) x 2(N); block tile 128x128
//  - wave tile 32x64: 2 M-frags x 4 N-frags, 8 f32 accumulators
//  - k-loop unrolled by 2 with fixed ping-pong fragment sets (no register
//    copies at the buffer hand-off); loads issue right after the WMMA octet
//    that frees their registers, giving ~8 WMMAs of latency cover and
//    partial s_wait_loadcnt overlap. Requires K % 64 == 0 (512/1024 here).
//  - two weight pointers: rows >= Mhalf use W1 (fwd/bwd batched GEMM)
//  - partial N (N=2192=16*137): out-of-range 16-col tiles clamp their B row
//    to N-1 (in-bounds, wave-uniform) and are skipped in the epilogue
//  - optional residual (may alias C: single read before the single write)
// Fragment lane layout per CDNA5 ISA 7.12.2 (16-bit A 16x32):
//   lanes 0-15 : row = lane,    K = {k0..k0+7, k0+16..k0+23}
//   lanes16-31 : row = lane-16, K = {k0+8..k0+15, k0+24..k0+31}
// C/D layout: VGPR i -> (M = i + 8*half, N = lane&15).
// -------------------------------------------------------------------------
__device__ __forceinline__ v16bf load_frag_p(const bf16_t* __restrict__ p) {
    union { v16bf v; v8bf h[2]; } u;
    u.h[0] = *(const v8bf*)(p);
    u.h[1] = *(const v8bf*)(p + 16);
    return u.v;
}

__device__ __forceinline__ void load_set(v16bf (&a)[2], v16bf (&b)[4],
                                         const bf16_t* const (&aP)[2],
                                         const bf16_t* const (&bP)[4], int k0) {
#pragma unroll
    for (int mi = 0; mi < 2; ++mi) a[mi] = load_frag_p(aP[mi] + k0);
#pragma unroll
    for (int j = 0; j < 4; ++j) b[j] = load_frag_p(bP[j] + k0);
}

__device__ __forceinline__ void wmma8(v8f (&acc)[2][4], const v16bf (&a)[2],
                                      const v16bf (&b)[4]) {
#pragma unroll
    for (int j = 0; j < 4; ++j) {
        acc[0][j] = __builtin_amdgcn_wmma_f32_16x16x32_bf16(
            false, a[0], false, b[j], (short)0, acc[0][j], false, false);
        acc[1][j] = __builtin_amdgcn_wmma_f32_16x16x32_bf16(
            false, a[1], false, b[j], (short)0, acc[1][j], false, false);
    }
}

__global__ void __launch_bounds__(256, 2)
k_gemm_bf16_wmma(const bf16_t* __restrict__ A,
                 const bf16_t* __restrict__ W0,
                 const bf16_t* __restrict__ W1,
                 const float* __restrict__ Res, float* __restrict__ C,
                 int M, int N, int K, int Mhalf) {
    const int wave = threadIdx.x >> 5;
    const int lane = threadIdx.x & 31;
    const int half = lane >> 4;
    const int r    = lane & 15;
    const int mA   = blockIdx.x * 128 + (wave >> 1) * 32;   // wave's first M-frag row
    const int nB   = blockIdx.y * 128 + (wave & 1) * 64;    // wave's first N-frag col
    const bf16_t* W = (W1 && mA >= Mhalf) ? W1 : W0;        // wave-uniform select

    // hoisted per-lane row base pointers (k folds into load offsets)
    const bf16_t* aP[2];
    aP[0] = A + (size_t)(mA + r) * K + half * 8;
    aP[1] = A + (size_t)(mA + 16 + r) * K + half * 8;
    const bf16_t* bP[4];
#pragma unroll
    for (int j = 0; j < 4; ++j) {
        int nrow = nB + j * 16 + r;
        nrow = nrow < N ? nrow : N - 1;                     // clamp (uniform per tile)
        bP[j] = W + (size_t)nrow * K + half * 8;
    }

    v8f acc[2][4];
#pragma unroll
    for (int mi = 0; mi < 2; ++mi)
#pragma unroll
        for (int j = 0; j < 4; ++j) acc[mi][j] = (v8f){};

    // ping-pong fragment sets with fixed roles (K % 64 == 0, K >= 64)
    v16bf aA[2], bA[4], aB[2], bB[4];
    load_set(aA, bA, aP, bP, 0);
    load_set(aB, bB, aP, bP, 32);
    for (int k0 = 64; k0 < K; k0 += 64) {
        wmma8(acc, aA, bA);
        load_set(aA, bA, aP, bP, k0);
        wmma8(acc, aB, bB);
        load_set(aB, bB, aP, bP, k0 + 32);
    }
    wmma8(acc, aA, bA);
    wmma8(acc, aB, bB);

#pragma unroll
    for (int mi = 0; mi < 2; ++mi)
#pragma unroll
        for (int j = 0; j < 4; ++j) {
            int col = nB + j * 16 + r;
            if (col >= N) continue;                         // uniform: N % 16 == 0
#pragma unroll
            for (int i = 0; i < 8; ++i) {
                int row = mA + mi * 16 + half * 8 + i;
                float v = acc[mi][j][i];
                if (Res) v += Res[(size_t)row * N + col];
                C[(size_t)row * N + col] = v;
            }
        }
}

// -------------------------------------------------------------------------
// depthwise causal conv (k=4) over zx[:,1024:2176] + bias + SiLU -> xBC
// both directions; per-direction weights selected by row block
// -------------------------------------------------------------------------
__global__ void k_conv_silu(const float* __restrict__ zx,
                            const float* __restrict__ cw0, const float* __restrict__ cb0,
                            const float* __restrict__ cw1, const float* __restrict__ cb1,
                            float* __restrict__ xBC) {
    int idx = blockIdx.x * 256 + threadIdx.x;
    if (idx >= L2 * CONVDIM) return;
    int tg = idx / CONVDIM, c = idx % CONVDIM;
    int dir = tg >> 10, t = tg & (LSEQ - 1);
    const float* cw = dir ? cw1 : cw0;
    const float* cb = dir ? cb1 : cb0;
    float acc = cb[c];
#pragma unroll
    for (int j = 0; j < DCONV; ++j) {
        int tt = t - (DCONV - 1) + j;
        if (tt >= 0)
            acc += cw[c * DCONV + j] * zx[(size_t)(dir * LSEQ + tt) * DINPROJ + DINNER + c];
    }
    xBC[idx] = silu_f(acc);
}

// -------------------------------------------------------------------------
// dt = softplus(dt_raw + bias); dA = exp(-exp(A_log) * dt); both directions
// -------------------------------------------------------------------------
__global__ void k_dtdA(const float* __restrict__ zx,
                       const float* __restrict__ dtb0, const float* __restrict__ Al0,
                       const float* __restrict__ dtb1, const float* __restrict__ Al1,
                       float* __restrict__ dt, float* __restrict__ dA) {
    int i = blockIdx.x * 256 + threadIdx.x;
    if (i >= L2 * NHEADS) return;
    int tg = i >> 4, h = i & 15;
    int dir = tg >> 10;
    const float* dtb = dir ? dtb1 : dtb0;
    const float* Al  = dir ? Al1  : Al0;
    float v = zx[(size_t)tg * DINPROJ + (DINPROJ - NHEADS) + h] + dtb[h];
    float sp = (v > 20.f) ? v : log1pf(__expf(v));
    dt[i] = sp;
    dA[i] = __expf(-__expf(Al[h]) * sp);
}

// -------------------------------------------------------------------------
// selective scan: 32 blocks = 2 directions x 16 heads, 64 lanes each.
// Each lane keeps its 64-entry state row in registers (64 VGPRs); B/C
// broadcast through LDS per step. Sequential over L (inherent recurrence).
// -------------------------------------------------------------------------
__global__ void k_scan(const float* __restrict__ xBC, const float* __restrict__ dt,
                       const float* __restrict__ dA,
                       const float* __restrict__ Dp0, const float* __restrict__ Dp1,
                       float* __restrict__ y) {
    const int dir = blockIdx.x >> 4;
    const int h   = blockIdx.x & 15;
    const int p   = threadIdx.x;         // 0..63
    const float* Dp = dir ? Dp1 : Dp0;
    float st[DSTATE];
#pragma unroll
    for (int n = 0; n < DSTATE; ++n) st[n] = 0.f;
    __shared__ float sB[DSTATE], sC[DSTATE];
    const float Dh = Dp[h];
    const int base = dir * LSEQ;
    for (int t = 0; t < LSEQ; ++t) {
        const float* row = xBC + (size_t)(base + t) * CONVDIM;
        sB[p] = row[DINNER + p];
        sC[p] = row[DINNER + DSTATE + p];
        __syncthreads();
        float x   = row[h * HEADDIM + p];
        float u   = dt[(base + t) * NHEADS + h] * x;
        float dAv = dA[(base + t) * NHEADS + h];
        float acc = 0.f;
#pragma unroll
        for (int n = 0; n < DSTATE; ++n) {
            st[n] = fmaf(st[n], dAv, u * sB[n]);
            acc   = fmaf(st[n], sC[n], acc);
        }
        y[(size_t)(base + t) * DINNER + h * HEADDIM + p] = acc + Dh * x;
        __syncthreads();
    }
}

// -------------------------------------------------------------------------
// gated RMSNorm: yg = y*silu(z); o = bf16(yg * rsqrt(mean(yg^2)+eps) * w)
// one block per row (2048 rows); per-direction norm weight
// -------------------------------------------------------------------------
__global__ void k_gated_rmsnorm(const float* __restrict__ y, const float* __restrict__ zx,
                                const float* __restrict__ nw0, const float* __restrict__ nw1,
                                bf16_t* __restrict__ o) {
    int t = blockIdx.x;
    const float* nw = (t >= LSEQ) ? nw1 : nw0;
    const float* yr = y + (size_t)t * DINNER;
    const float* zr = zx + (size_t)t * DINPROJ;   // z = first DINNER cols
    __shared__ float sred[9];
    float vals[4];
    float s2 = 0.f;
#pragma unroll
    for (int k = 0; k < 4; ++k) {
        int i = threadIdx.x + k * 256;
        float g = yr[i] * silu_f(zr[i]);
        vals[k] = g;
        s2 += g * g;
    }
    for (int off = 16; off; off >>= 1) s2 += __shfl_down(s2, off, 32);
    if ((threadIdx.x & 31) == 0) sred[threadIdx.x >> 5] = s2;
    __syncthreads();
    if (threadIdx.x == 0) {
        float a = 0.f;
        for (int i = 0; i < 8; ++i) a += sred[i];
        sred[8] = rsqrtf(a / (float)DINNER + EPSV);
    }
    __syncthreads();
    float scale = sred[8];
#pragma unroll
    for (int k = 0; k < 4; ++k) {
        int i = threadIdx.x + k * 256;
        o[(size_t)t * DINNER + i] = (bf16_t)(vals[k] * scale * nw[i]);
    }
}

// -------------------------------------------------------------------------
// concat [fwd(t), bwd(L-1-t)] -> bf16 (un-reverses the bwd stream)
// -------------------------------------------------------------------------
__global__ void k_concat_bf16(const float* __restrict__ xall, bf16_t* __restrict__ o) {
    int i = blockIdx.x * 256 + threadIdx.x;
    if (i >= LSEQ * 2 * DMODEL) return;
    int t = i >> 10, c = i & 1023;
    float v = (c < DMODEL)
        ? xall[(size_t)t * DMODEL + c]
        : xall[(size_t)(LSEQ + (LSEQ - 1 - t)) * DMODEL + (c - DMODEL)];
    o[i] = (bf16_t)v;
}

// -------------------------------------------------------------------------
// host-side orchestration (all launches on `stream`; graph-capture safe)
// -------------------------------------------------------------------------
static inline char* ws_take(char*& cur, size_t bytes) {
    char* p = cur;
    cur += (bytes + 255) & ~size_t(255);
    return p;
}

static inline void launch_f2bf(const float* s, bf16_t* d, int n, hipStream_t stream) {
    int nv = (n + 3) / 4;
    k_f2bf<<<(nv + 255) / 256, 256, 0, stream>>>(s, d, n);
}

extern "C" void kernel_launch(void* const* d_in, const int* in_sizes, int n_in,
                              void* d_out, int out_size, void* d_ws, size_t ws_size,
                              hipStream_t stream) {
    (void)in_sizes; (void)n_in; (void)out_size; (void)ws_size;
    const int*   tokens  = (const int*)d_in[0];
    const float* embed_W = (const float*)d_in[1];
    const float* merge_W = (const float*)d_in[22];
    const float* fln_w   = (const float*)d_in[23];
    const float* fln_b   = (const float*)d_in[24];
    // per-direction parameter bases: fwd = d_in[2..11], bwd = d_in[12..21]
    const float* P[2][10];
    for (int dir = 0; dir < 2; ++dir)
        for (int k = 0; k < 10; ++k)
            P[dir][k] = (const float*)d_in[2 + dir * 10 + k];

    char* cur = (char*)d_ws;
    float*  xall = (float*)ws_take(cur, (size_t)L2 * DMODEL * 4);   // fwd rows then bwd rows
    bf16_t* xnbf = (bf16_t*)ws_take(cur, (size_t)L2 * DMODEL * 2);
    float*  zx   = (float*)ws_take(cur, (size_t)L2 * DINPROJ * 4);
    float*  xBC  = (float*)ws_take(cur, (size_t)L2 * CONVDIM * 4);
    float*  dtb  = (float*)ws_take(cur, (size_t)L2 * NHEADS * 4);
    float*  dAb  = (float*)ws_take(cur, (size_t)L2 * NHEADS * 4);
    float*  ybuf = (float*)ws_take(cur, (size_t)L2 * DINNER * 4);
    bf16_t* ybf  = (bf16_t*)ws_take(cur, (size_t)L2 * DINNER * 2);
    bf16_t* mbf  = (bf16_t*)ws_take(cur, (size_t)LSEQ * 2 * DMODEL * 2);
    bf16_t* fbf  = (bf16_t*)ws_take(cur, (size_t)LSEQ * DMODEL * 2);
    bf16_t* wbf  = (bf16_t*)ws_take(cur, (size_t)VOCAB * DMODEL * 2); // biggest weight

    // embeddings (fwd + time-reversed bwd in one buffer)
    k_embed<<<LSEQ, DMODEL, 0, stream>>>(tokens, embed_W, xall);

    for (int l = 0; l < DEPTH; ++l) {
        const float* in_proj[2];
        const float* out_proj[2];
        for (int d = 0; d < 2; ++d) {
            in_proj[d]  = P[d][0] + (size_t)l * DINPROJ * DMODEL;
            out_proj[d] = P[d][7] + (size_t)l * DMODEL * DINNER;
        }
        const float* cw[2]  = {P[0][1] + (size_t)l * CONVDIM * DCONV, P[1][1] + (size_t)l * CONVDIM * DCONV};
        const float* cb[2]  = {P[0][2] + (size_t)l * CONVDIM,         P[1][2] + (size_t)l * CONVDIM};
        const float* db[2]  = {P[0][3] + (size_t)l * NHEADS,          P[1][3] + (size_t)l * NHEADS};
        const float* Al[2]  = {P[0][4] + (size_t)l * NHEADS,          P[1][4] + (size_t)l * NHEADS};
        const float* Dp[2]  = {P[0][5] + (size_t)l * NHEADS,          P[1][5] + (size_t)l * NHEADS};
        const float* nw[2]  = {P[0][6] + (size_t)l * DINNER,          P[1][6] + (size_t)l * DINNER};
        const float* lw[2]  = {P[0][8] + (size_t)l * DMODEL,          P[1][8] + (size_t)l * DMODEL};
        const float* lb[2]  = {P[0][9] + (size_t)l * DMODEL,          P[1][9] + (size_t)l * DMODEL};

        // layernorm (both dirs) -> bf16
        k_layernorm_bf16<<<L2, 256, 0, stream>>>(xall, lw[0], lb[0], lw[1], lb[1], xnbf, DMODEL);
        // in_proj GEMM (M=2048, N=2192, K=512), batched fwd/bwd weights
        const int ipn = DINPROJ * DMODEL;
        launch_f2bf(in_proj[0], wbf,       ipn, stream);
        launch_f2bf(in_proj[1], wbf + ipn, ipn, stream);
        k_gemm_bf16_wmma<<<dim3(L2 / 128, (DINPROJ + 127) / 128), 256, 0, stream>>>(
            xnbf, wbf, wbf + ipn, nullptr, zx, L2, DINPROJ, DMODEL, LSEQ);
        // conv + silu
        k_conv_silu<<<(L2 * CONVDIM + 255) / 256, 256, 0, stream>>>(
            zx, cw[0], cb[0], cw[1], cb[1], xBC);
        // dt / dA
        k_dtdA<<<(L2 * NHEADS + 255) / 256, 256, 0, stream>>>(
            zx, db[0], Al[0], db[1], Al[1], dtb, dAb);
        // sequential scan (2 dirs x 16 heads)
        k_scan<<<2 * NHEADS, HEADDIM, 0, stream>>>(xBC, dtb, dAb, Dp[0], Dp[1], ybuf);
        // gated RMSNorm -> bf16
        k_gated_rmsnorm<<<L2, 256, 0, stream>>>(ybuf, zx, nw[0], nw[1], ybf);
        // out_proj GEMM with fused residual (M=2048, N=512, K=1024)
        const int opn = DMODEL * DINNER;
        launch_f2bf(out_proj[0], wbf,       opn, stream);
        launch_f2bf(out_proj[1], wbf + opn, opn, stream);
        k_gemm_bf16_wmma<<<dim3(L2 / 128, DMODEL / 128), 256, 0, stream>>>(
            ybf, wbf, wbf + opn, xall, xall, L2, DMODEL, DINNER, LSEQ);
    }

    // concat fwd + un-reversed bwd -> bf16
    k_concat_bf16<<<(LSEQ * 2 * DMODEL + 255) / 256, 256, 0, stream>>>(xall, mbf);
    // merge GEMM (M=1024, N=512, K=1024) -> reuse ybuf for f32 merged output
    const int mwn = DMODEL * 2 * DMODEL;
    launch_f2bf(merge_W, wbf, mwn, stream);
    k_gemm_bf16_wmma<<<dim3(LSEQ / 128, DMODEL / 128), 256, 0, stream>>>(
        mbf, wbf, nullptr, nullptr, ybuf, LSEQ, DMODEL, 2 * DMODEL, LSEQ);
    // final layernorm -> bf16
    k_layernorm_bf16<<<LSEQ, 256, 0, stream>>>(ybuf, fln_w, fln_b, fln_w, fln_b, fbf, DMODEL);
    // logits GEMM (M=1024, N=32000, K=512) -> d_out (f32)
    const int ewn = VOCAB * DMODEL;
    launch_f2bf(embed_W, wbf, ewn, stream);
    k_gemm_bf16_wmma<<<dim3(LSEQ / 128, VOCAB / 128), 256, 0, stream>>>(
        fbf, wbf, nullptr, nullptr, (float*)d_out, LSEQ, VOCAB, DMODEL, LSEQ);
}
